// LongformerEncoderLayer_89696097009904
// MI455X (gfx1250) — matrix operand order repaired and checked by reading, not compiled, not run on hardware
//
#include <hip/hip_runtime.h>
#include <hip/hip_bf16.h>

typedef __attribute__((ext_vector_type(16))) _Float16 v16h;
typedef __attribute__((ext_vector_type(8)))  _Float16 v8h;
typedef __attribute__((ext_vector_type(8)))  float    v8f;

#define DEV static __device__ __forceinline__

constexpr int kB = 2, kS = 4096, kDM = 768, kDFF = 3072;
constexpr int kH = 12, kD = 64, kW = 128, kG = 64, kNC = kS / kW;
constexpr int kL = 3 * kW + kG;        // 448 score columns (band + global)
constexpr float kNEG = -1.0e9f;
constexpr float kEPS = 1e-5f;
constexpr float kSCALE = 0.125f;       // 1/sqrt(64)

// ---------------------------------------------------------------------------
// CDNA5 async global -> LDS copy (GLOBAL_LOAD_ASYNC_TO_LDS_B128, ASYNCcnt).
// LDS operand: low 32 bits of the flat shared-aperture address == LDS offset
// (ISA 10.2).
// ---------------------------------------------------------------------------
DEV void async_copy_b128(void* ldsDst, const void* gSrc) {
  const unsigned loff = (unsigned)(size_t)ldsDst;
  asm volatile("global_load_async_to_lds_b128 %0, %1, off"
               :
               : "v"(loff), "v"(gSrc)
               : "memory");
}

DEV void async_wait() {
#if __has_builtin(__builtin_amdgcn_s_wait_asynccnt)
  __builtin_amdgcn_s_wait_asynccnt(0);
#else
  asm volatile("s_wait_asynccnt 0x0" ::: "memory");
#endif
}

// ---------------------------------------------------------------------------
// WMMA fragment helpers (layouts per CDNA5 ISA 7.12.2, wave32)
// ---------------------------------------------------------------------------

// A-matrix 16x32 f16: lane = M (mod 16); VGPR v holds K pair
//   K = ((v>=4)?16:0) + ((lane>=16)?8:0) + 2*(v%4) + {0,1}
DEV v16h frag_a(const _Float16* base, int ld) {
  const int lane = threadIdx.x & 31;
  const _Float16* row = base + (size_t)(lane & 15) * ld;
  const int kg = (lane & 16) >> 1;   // 0 or 8
  v16h f;
#pragma unroll
  for (int v = 0; v < 8; ++v) {
    const int k = ((v & 4) << 2) + kg + ((v & 3) << 1);
    f[2 * v]     = row[k];
    f[2 * v + 1] = row[k + 1];
  }
  return f;
}

// B-matrix 32x16 f16: lane = N (mod 16); element s holds
//   K = ((s>=8)?16:0) + ((lane>=16)?8:0) + (s%8)
DEV v16h frag_b(const _Float16* base, int ld) {
  const int lane = threadIdx.x & 31;
  const _Float16* col = base + (lane & 15);
  const int kg = (lane & 16) >> 1;   // 0 or 8
  v16h f;
#pragma unroll
  for (int s = 0; s < 16; ++s) {
    const int k = ((s & 8) << 1) + kg + (s & 7);
    f[s] = col[(size_t)k * ld];
  }
  return f;
}

DEV v8f wmma32(v16h a, v16h b, v8f c) {
  return __builtin_amdgcn_wmma_f32_16x16x32_f16(false, a, false, b, (short)0, c,
                                                false, false);
}

// ---------------------------------------------------------------------------
// f32 -> f16 conversion, 8 elements per thread iteration (b128 traffic)
// ---------------------------------------------------------------------------
__global__ void cvt_kernel(const float* __restrict__ src,
                           _Float16* __restrict__ dst, size_t n8) {
  size_t i = (size_t)blockIdx.x * blockDim.x + threadIdx.x;
  const size_t stride = (size_t)gridDim.x * blockDim.x;
  const float4* s4 = (const float4*)src;
  v8h* d8 = (v8h*)dst;
  for (; i < n8; i += stride) {
    const float4 a = s4[2 * i];
    const float4 b = s4[2 * i + 1];
    v8h o;
    o[0] = (_Float16)a.x; o[1] = (_Float16)a.y;
    o[2] = (_Float16)a.z; o[3] = (_Float16)a.w;
    o[4] = (_Float16)b.x; o[5] = (_Float16)b.y;
    o[6] = (_Float16)b.z; o[7] = (_Float16)b.w;
    d8[i] = o;
  }
}

// ---------------------------------------------------------------------------
// global token index selection: stable argsort(!gmask)[:G]
// ---------------------------------------------------------------------------
__global__ void gidx_kernel(const unsigned char* __restrict__ gmask,
                            int* __restrict__ gidx) {
  const int b = blockIdx.x;
  if (threadIdx.x != 0) return;
  int cnt = 0;
  for (int s = 0; s < kS && cnt < kG; ++s)
    if (gmask[(size_t)b * kS + s]) gidx[b * kG + cnt++] = s;
  for (int s = 0; s < kS && cnt < kG; ++s)
    if (!gmask[(size_t)b * kS + s]) gidx[b * kG + cnt++] = s;
}

// ---------------------------------------------------------------------------
// Tiled f16 WMMA GEMM:  C[M,N] = act( (A[M,K] @ B[K,N] + bias) * alpha )
// block = 128 threads (4 waves, 2x2 grid of 32x32 wave tiles), tile 64x64x32
// Staging via async global->LDS b128; next-tile prefetch (global_prefetch_b8).
// ---------------------------------------------------------------------------
__global__ __launch_bounds__(128) void gemm_f16_kernel(
    const _Float16* __restrict__ A, const _Float16* __restrict__ B,
    const float* __restrict__ bias, float* __restrict__ Cf,
    _Float16* __restrict__ Ch, int M, int N, int K, float alpha, int relu) {
  constexpr int BM = 64, BN = 64, BK = 32;
  __shared__ __align__(16) _Float16 As[BM * BK];
  __shared__ __align__(16) _Float16 Bs[BK * BN];

  const int tid = threadIdx.x;
  const int wave = tid >> 5, lane = tid & 31;
  const int wm = wave >> 1, wn = wave & 1;
  const int m0 = blockIdx.y * BM, n0 = blockIdx.x * BN;

  v8f acc[2][2] = {};

  for (int k0 = 0; k0 < K; k0 += BK) {
    // A tile: 64 rows x 32 halfs = 256 x 16B async chunks (4 per row)
    for (int ch = tid; ch < BM * BK / 8; ch += 128) {
      const int m = ch >> 2, k8 = (ch & 3) << 3;
      async_copy_b128(As + m * BK + k8,
                      A + (size_t)(m0 + m) * K + k0 + k8);
    }
    // B tile: 32 rows x 64 halfs = 256 x 16B async chunks (8 per row)
    for (int ch = tid; ch < BK * BN / 8; ch += 128) {
      const int kk = ch >> 3, n8 = (ch & 7) << 3;
      async_copy_b128(Bs + kk * BN + n8,
                      B + (size_t)(k0 + kk) * N + n0 + n8);
    }
    if (k0 + BK < K) {  // prefetch next K tile (global_prefetch_b8)
      __builtin_prefetch(A + (size_t)(m0 + (tid >> 1)) * K + k0 + BK, 0, 3);
      __builtin_prefetch(B + (size_t)(k0 + BK + (tid >> 2)) * N + n0, 0, 3);
    }
    async_wait();
    __syncthreads();

    const v16h a0 = frag_a(As + (wm * 32) * BK, BK);
    const v16h a1 = frag_a(As + (wm * 32 + 16) * BK, BK);
    const v16h b0 = frag_b(Bs + wn * 32, BN);
    const v16h b1 = frag_b(Bs + wn * 32 + 16, BN);
    acc[0][0] = wmma32(a0, b0, acc[0][0]);
    acc[0][1] = wmma32(a0, b1, acc[0][1]);
    acc[1][0] = wmma32(a1, b0, acc[1][0]);
    acc[1][1] = wmma32(a1, b1, acc[1][1]);
    __syncthreads();
  }

  const int mo = (lane >> 4) << 3, nn = lane & 15;
#pragma unroll
  for (int i = 0; i < 2; ++i)
#pragma unroll
    for (int j = 0; j < 2; ++j)
#pragma unroll
      for (int r = 0; r < 8; ++r) {
        const int m = m0 + wm * 32 + i * 16 + r + mo;
        const int n = n0 + wn * 32 + j * 16 + nn;
        float v = (acc[i][j][r] + bias[n]) * alpha;
        if (relu) v = fmaxf(v, 0.0f);
        const size_t o = (size_t)m * N + n;
        if (Cf) Cf[o] = v;
        if (Ch) Ch[o] = (_Float16)v;
      }
}

// ---------------------------------------------------------------------------
// Fused band + global-column attention.
// One 64-thread (2-wave) WG handles 32 query rows of one (b, h, chunk).
// Score tiles computed TRANSPOSED (keys = M dim, queries = N dim) so K and V
// both stage row-major [key][d] with async b128 copies.
// Dynamic LDS: Sb f32[32][448] | KV f16[448][64] (K then V) |
//              Pb f16[32][448] | Qt f16[64][32] | scol i32[448]
// ---------------------------------------------------------------------------
__global__ __launch_bounds__(64) void band_attn_kernel(
    const _Float16* __restrict__ qh, const _Float16* __restrict__ kh,
    const _Float16* __restrict__ vh, const unsigned char* __restrict__ pad,
    const unsigned char* __restrict__ gmask, const int* __restrict__ gidx,
    float* __restrict__ attn) {
  extern __shared__ __align__(16) char smem[];
  float*    Sb   = (float*)smem;                                   // 57344 B
  _Float16* KV   = (_Float16*)(smem + 32 * kL * 4);                // 57344 B
  _Float16* Pb   = (_Float16*)(smem + 32 * kL * 4 + 64 * kL * 2);  // 28672 B
  _Float16* Qt   = (_Float16*)(smem + 32 * kL * 4 + 64 * kL * 2 + 32 * kL * 2);
  int* scol = (int*)(smem + 32 * kL * 4 + 64 * kL * 2 + 32 * kL * 2 + 64 * 32 * 2);

  int bid = blockIdx.x;
  const int sub = bid & 3;          bid >>= 2;
  const int c   = bid & (kNC - 1);  bid >>= 5;
  const int h   = bid % kH;
  const int b   = bid / kH;
  const int tid = threadIdx.x, wave = tid >> 5, lane = tid & 31;
  const int mo = (lane >> 4) << 3, nn = lane & 15;

  // column -> sequence index (-1 if masked)
  for (int j = tid; j < kL; j += 64) {
    int s; bool ok;
    if (j < 3 * kW) {
      s = c * kW + j - kW;
      ok = (s >= 0) && (s < kS) && pad[(size_t)b * kS + s] &&
           !gmask[(size_t)b * kS + s];
    } else {
      s = gidx[b * kG + (j - 3 * kW)];
      ok = pad[(size_t)b * kS + s] != 0;
    }
    scol[j] = ok ? s : -1;
  }
  // stage Q transposed: Qt[d][row], rows are this WG's 32 queries
  const int sq_base = c * kW + sub * 32;
  for (int ch = tid; ch < 32 * kD / 8; ch += 64) {
    const int row = ch >> 3, d0 = (ch & 7) << 3;
    const v8h val = *(const v8h*)(qh + ((size_t)b * kS + sq_base + row) * kDM +
                                  h * kD + d0);
#pragma unroll
    for (int t = 0; t < 8; ++t) Qt[(d0 + t) * 32 + row] = val[t];
  }
  __syncthreads();

  // stage K window row-major [key][d] : async b128; masked rows zero-filled
  for (int ch = tid; ch < kL * kD / 8; ch += 64) {
    const int j = ch >> 3, d0 = (ch & 7) << 3;
    const int s = scol[j];
    if (s >= 0) {
      async_copy_b128(KV + j * kD + d0,
                      kh + ((size_t)b * kS + s) * kDM + h * kD + d0);
    } else {
      const v8h z = {};
      *(v8h*)(KV + j * kD + d0) = z;
    }
  }
  async_wait();
  __syncthreads();

  // q-side B fragments (constant for the whole score pass)
  v16h bq[2][2];
#pragma unroll
  for (int kk = 0; kk < 2; ++kk)
#pragma unroll
    for (int nt = 0; nt < 2; ++nt)
      bq[kk][nt] = frag_b(Qt + kk * 32 * 32 + nt * 16, 32);

  // transposed scores: S^T[j][q]; each wave owns 14 of the 28 j-tiles
  for (int jt = wave * 14; jt < wave * 14 + 14; ++jt) {
    const v16h a0 = frag_a(KV + (jt * 16) * kD + 0, kD);
    const v16h a1 = frag_a(KV + (jt * 16) * kD + 32, kD);
#pragma unroll
    for (int nt = 0; nt < 2; ++nt) {
      v8f acc = {};
      acc = wmma32(a0, bq[0][nt], acc);
      acc = wmma32(a1, bq[1][nt], acc);
#pragma unroll
      for (int r = 0; r < 8; ++r) {
        const int j = jt * 16 + r + mo;        // key column
        const int qrow = nt * 16 + nn;         // query row (0..31)
        const int qq = sub * 32 + qrow;        // row within chunk
        const bool ok =
            (scol[j] >= 0) &&
            ((j >= 3 * kW) || ((j >= qq) && (j <= qq + 2 * kW)));
        Sb[qrow * kL + j] = ok ? acc[r] : kNEG;
      }
    }
  }
  __syncthreads();

  // softmax over 448 columns, write probabilities (f16) row-major [q][key]
  if (tid < 32) {
    const float* srow = Sb + tid * kL;
    _Float16* prow = Pb + tid * kL;
    float mx = kNEG;
    for (int j = 0; j < kL; ++j) mx = fmaxf(mx, srow[j]);
    float sum = 0.0f;
    for (int j = 0; j < kL; ++j) sum += __expf(srow[j] - mx);
    const float inv = 1.0f / sum;
    for (int j = 0; j < kL; ++j)
      prow[j] = (_Float16)(__expf(srow[j] - mx) * inv);
  }
  __syncthreads();

  // restage V window row-major [key][d] into KV (async b128)
  for (int ch = tid; ch < kL * kD / 8; ch += 64) {
    const int j = ch >> 3, d0 = (ch & 7) << 3;
    const int s = scol[j];
    if (s >= 0) {
      async_copy_b128(KV + j * kD + d0,
                      vh + ((size_t)b * kS + s) * kDM + h * kD + d0);
    } else {
      const v8h z = {};
      *(v8h*)(KV + j * kD + d0) = z;
    }
  }
  async_wait();
  __syncthreads();

  // out = P @ V : 16 x 64 per wave
  const int sq0 = sq_base + wave * 16;
#pragma unroll
  for (int nt = 0; nt < 4; ++nt) {
    v8f acc = {};
    for (int kt = 0; kt < kL / 32; ++kt) {
      const v16h a  = frag_a(Pb + (wave * 16) * kL + kt * 32, kL);
      const v16h bb = frag_b(KV + (kt * 32) * kD + nt * 16, kD);
      acc = wmma32(a, bb, acc);
    }
#pragma unroll
    for (int r = 0; r < 8; ++r) {
      const int m = r + mo;
      attn[((size_t)b * kS + sq0 + m) * kDM + h * kD + nt * 16 + nn] = acc[r];
    }
  }
}

// ---------------------------------------------------------------------------
// Global-query attention: qg (64 rows) attends all S keys, two-pass softmax.
// Transposed score tiles (keys = M dim); K/V tiles staged async row-major.
// One 128-thread (4-wave) WG per (b, h); overwrites attn rows at gidx.
// ---------------------------------------------------------------------------
__global__ __launch_bounds__(128) void global_attn_kernel(
    const _Float16* __restrict__ qh, const _Float16* __restrict__ kh,
    const _Float16* __restrict__ vh, const unsigned char* __restrict__ pad,
    const int* __restrict__ gidx, float* __restrict__ attn) {
  __shared__ __align__(16) _Float16 Qt[kD * kG];   // transposed [d][g]
  __shared__ __align__(16) _Float16 Ks[64 * kD];   // [key][d]
  __shared__ __align__(16) _Float16 Vs[64 * kD];   // [key][d]
  __shared__ __align__(16) _Float16 Ps[kG * 64];   // [g][key]
  __shared__ float St[64 * kG];                    // S^T [key][g]
  __shared__ float rowM[kG], rowS[kG];
  __shared__ int   padok[64];

  const int h = blockIdx.x % kH, b = blockIdx.x / kH;
  const int tid = threadIdx.x, wave = tid >> 5, lane = tid & 31;
  const int mo = (lane >> 4) << 3, nn = lane & 15;

  // stage global queries transposed: Qt[d][g]
  for (int ch = tid; ch < kG * kD / 8; ch += 128) {
    const int g = ch >> 3, d0 = (ch & 7) << 3;
    const int s = gidx[b * kG + g];
    const v8h val =
        *(const v8h*)(qh + ((size_t)b * kS + s) * kDM + h * kD + d0);
#pragma unroll
    for (int t = 0; t < 8; ++t) Qt[(d0 + t) * kG + g] = val[t];
  }
  if (tid < kG) { rowM[tid] = kNEG; rowS[tid] = 0.0f; }
  __syncthreads();

  // q-side B fragments (constant across the whole kernel)
  v16h bq[2][4];
#pragma unroll
  for (int kk = 0; kk < 2; ++kk)
#pragma unroll
    for (int nt = 0; nt < 4; ++nt)
      bq[kk][nt] = frag_b(Qt + kk * 32 * kG + nt * 16, kG);

  // ---- pass 1: running row max / sum ----
  for (int s0 = 0; s0 < kS; s0 += 64) {
    for (int ch = tid; ch < 64 * kD / 8; ch += 128) {
      const int j = ch >> 3, d0 = (ch & 7) << 3;
      async_copy_b128(Ks + j * kD + d0,
                      kh + ((size_t)b * kS + s0 + j) * kDM + h * kD + d0);
    }
    if (tid < 64) padok[tid] = pad[(size_t)b * kS + s0 + tid];
    async_wait();
    __syncthreads();

    const v16h a0 = frag_a(Ks + (wave * 16) * kD + 0, kD);
    const v16h a1 = frag_a(Ks + (wave * 16) * kD + 32, kD);
#pragma unroll
    for (int nt = 0; nt < 4; ++nt) {
      v8f acc = {};
      acc = wmma32(a0, bq[0][nt], acc);
      acc = wmma32(a1, bq[1][nt], acc);
#pragma unroll
      for (int r = 0; r < 8; ++r) {
        const int j = wave * 16 + r + mo;      // key within tile
        const int g = nt * 16 + nn;            // query row
        St[j * kG + g] = padok[j] ? acc[r] : kNEG;
      }
    }
    __syncthreads();
    if (tid < kG) {
      const int g = tid;
      float tmax = kNEG;
      for (int j = 0; j < 64; ++j) tmax = fmaxf(tmax, St[j * kG + g]);
      const float mOld = rowM[g];
      const float mNew = fmaxf(mOld, tmax);
      float tsum = 0.0f;
      for (int j = 0; j < 64; ++j) tsum += __expf(St[j * kG + g] - mNew);
      rowS[g] = rowS[g] * __expf(mOld - mNew) + tsum;
      rowM[g] = mNew;
    }
    __syncthreads();
  }

  // ---- pass 2: P @ V accumulate ----
  v8f o[4] = {};
  for (int s0 = 0; s0 < kS; s0 += 64) {
    for (int ch = tid; ch < 64 * kD / 8; ch += 128) {
      const int j = ch >> 3, d0 = (ch & 7) << 3;
      async_copy_b128(Ks + j * kD + d0,
                      kh + ((size_t)b * kS + s0 + j) * kDM + h * kD + d0);
      async_copy_b128(Vs + j * kD + d0,
                      vh + ((size_t)b * kS + s0 + j) * kDM + h * kD + d0);
    }
    if (tid < 64) padok[tid] = pad[(size_t)b * kS + s0 + tid];
    async_wait();
    __syncthreads();

    const v16h a0 = frag_a(Ks + (wave * 16) * kD + 0, kD);
    const v16h a1 = frag_a(Ks + (wave * 16) * kD + 32, kD);
#pragma unroll
    for (int nt = 0; nt < 4; ++nt) {
      v8f acc = {};
      acc = wmma32(a0, bq[0][nt], acc);
      acc = wmma32(a1, bq[1][nt], acc);
#pragma unroll
      for (int r = 0; r < 8; ++r) {
        const int j = wave * 16 + r + mo;
        const int g = nt * 16 + nn;
        St[j * kG + g] = padok[j] ? acc[r] : kNEG;
      }
    }
    __syncthreads();
    if (tid < kG) {
      const int g = tid;
      const float im = rowM[g];
      const float is = rowS[g] > 0.0f ? 1.0f / rowS[g] : 0.0f;
      for (int j = 0; j < 64; ++j)
        Ps[g * 64 + j] = (_Float16)(__expf(St[j * kG + g] - im) * is);
    }
    __syncthreads();
    const v16h ap0 = frag_a(Ps + (wave * 16) * 64, 64);
    const v16h ap1 = frag_a(Ps + (wave * 16) * 64 + 32, 64);
#pragma unroll
    for (int nt = 0; nt < 4; ++nt) {
      o[nt] = wmma32(ap0, frag_b(Vs + 0 * kD + nt * 16, kD), o[nt]);
      o[nt] = wmma32(ap1, frag_b(Vs + 32 * kD + nt * 16, kD), o[nt]);
    }
    __syncthreads();
  }

#pragma unroll
  for (int nt = 0; nt < 4; ++nt)
#pragma unroll
    for (int r = 0; r < 8; ++r) {
      const int g = wave * 16 + r + mo;
      const int s = gidx[b * kG + g];
      attn[((size_t)b * kS + s) * kDM + h * kD + nt * 16 + nn] = o[nt][r];
    }
}

// ---------------------------------------------------------------------------
// residual add + LayerNorm (one row per block), f32 out + optional f16 out
// ---------------------------------------------------------------------------
__global__ __launch_bounds__(256) void add_ln_kernel(
    const float* __restrict__ x, const float* __restrict__ y,
    const float* __restrict__ gam, const float* __restrict__ bet,
    float* __restrict__ outf, _Float16* __restrict__ outh) {
  __shared__ float red[256];
  const int row = blockIdx.x, tid = threadIdx.x;
  const float* xr = x + (size_t)row * kDM;
  const float* yr = y + (size_t)row * kDM;

  float s = 0.0f;
  for (int i = tid; i < kDM; i += 256) s += xr[i] + yr[i];
  red[tid] = s;
  __syncthreads();
  for (int o = 128; o > 0; o >>= 1) {
    if (tid < o) red[tid] += red[tid + o];
    __syncthreads();
  }
  const float mean = red[0] / (float)kDM;
  __syncthreads();

  float v = 0.0f;
  for (int i = tid; i < kDM; i += 256) {
    const float d = xr[i] + yr[i] - mean;
    v += d * d;
  }
  red[tid] = v;
  __syncthreads();
  for (int o = 128; o > 0; o >>= 1) {
    if (tid < o) red[tid] += red[tid + o];
    __syncthreads();
  }
  const float inv = rsqrtf(red[0] / (float)kDM + kEPS);

  for (int i = tid; i < kDM; i += 256) {
    const float o2 = (xr[i] + yr[i] - mean) * inv * gam[i] + bet[i];
    outf[(size_t)row * kDM + i] = o2;
    if (outh) outh[(size_t)row * kDM + i] = (_Float16)o2;
  }
}

// ---------------------------------------------------------------------------
// host-side pipeline
// ---------------------------------------------------------------------------
extern "C" void kernel_launch(void* const* d_in, const int* in_sizes, int n_in,
                              void* d_out, int out_size, void* d_ws,
                              size_t ws_size, hipStream_t stream) {
  (void)in_sizes; (void)n_in; (void)out_size; (void)ws_size;

  const float* x  = (const float*)d_in[0];
  const unsigned char* pad = (const unsigned char*)d_in[1];
  const unsigned char* gm  = (const unsigned char*)d_in[2];
  const float* Wq = (const float*)d_in[3];  const float* bq  = (const float*)d_in[4];
  const float* Wk = (const float*)d_in[5];  const float* bk  = (const float*)d_in[6];
  const float* Wv = (const float*)d_in[7];  const float* bv  = (const float*)d_in[8];
  const float* Wo = (const float*)d_in[9];  const float* bo  = (const float*)d_in[10];
  const float* W1 = (const float*)d_in[11]; const float* b1  = (const float*)d_in[12];
  const float* W2 = (const float*)d_in[13]; const float* b2  = (const float*)d_in[14];
  const float* g1 = (const float*)d_in[15]; const float* be1 = (const float*)d_in[16];
  const float* g2 = (const float*)d_in[17]; const float* be2 = (const float*)d_in[18];
  float* out = (float*)d_out;

  const size_t nBSdm = (size_t)kB * kS * kDM;
  const size_t nDM2  = (size_t)kDM * kDM;
  const size_t nDMFF = (size_t)kDM * kDFF;
  const size_t nBSff = (size_t)kB * kS * kDFF;

  char* ws = (char*)d_ws;
  size_t off = 0;
  auto take = [&](size_t bytes) {
    char* p = ws + off;
    off += (bytes + 255) & ~(size_t)255;
    return p;
  };

  _Float16* xh    = (_Float16*)take(nBSdm * 2);
  _Float16* wqh   = (_Float16*)take(nDM2 * 2);
  _Float16* wkh   = (_Float16*)take(nDM2 * 2);
  _Float16* wvh   = (_Float16*)take(nDM2 * 2);
  _Float16* woh   = (_Float16*)take(nDM2 * 2);
  _Float16* w1h   = (_Float16*)take(nDMFF * 2);
  _Float16* w2h   = (_Float16*)take(nDMFF * 2);
  _Float16* qhb   = (_Float16*)take(nBSdm * 2);
  _Float16* khb   = (_Float16*)take(nBSdm * 2);
  _Float16* vhb   = (_Float16*)take(nBSdm * 2);
  float*    attnf = (float*)   take(nBSdm * 4);
  _Float16* attnh = (_Float16*)take(nBSdm * 2);
  float*    of    = (float*)   take(nBSdm * 4);
  float*    h1f   = (float*)   take(nBSdm * 4);
  _Float16* h1h   = (_Float16*)take(nBSdm * 2);
  _Float16* ff1h  = (_Float16*)take(nBSff * 2);
  float*    ff2f  = (float*)   take(nBSdm * 4);
  int*      gidx  = (int*)     take((size_t)kB * kG * sizeof(int));

  auto cvt = [&](const float* src, _Float16* dst, size_t n) {
    const size_t n8 = n / 8;
    int blocks = (int)((n8 + 255) / 256);
    if (blocks > 2048) blocks = 2048;
    cvt_kernel<<<blocks, 256, 0, stream>>>(src, dst, n8);
  };

  // 1. precision conversion of activations + weights
  cvt(x,  xh,  nBSdm);
  cvt(Wq, wqh, nDM2);  cvt(Wk, wkh, nDM2);
  cvt(Wv, wvh, nDM2);  cvt(Wo, woh, nDM2);
  cvt(W1, w1h, nDMFF); cvt(W2, w2h, nDMFF);

  // 2. global token indices
  gidx_kernel<<<kB, 32, 0, stream>>>(gm, gidx);

  const int Mrows = kB * kS;
  const dim3 gP(kDM / 64, Mrows / 64);

  // 3. Q/K/V projections (Q folds in 1/sqrt(D))
  gemm_f16_kernel<<<gP, 128, 0, stream>>>(xh, wqh, bq, nullptr, qhb,
                                          Mrows, kDM, kDM, kSCALE, 0);
  gemm_f16_kernel<<<gP, 128, 0, stream>>>(xh, wkh, bk, nullptr, khb,
                                          Mrows, kDM, kDM, 1.0f, 0);
  gemm_f16_kernel<<<gP, 128, 0, stream>>>(xh, wvh, bv, nullptr, vhb,
                                          Mrows, kDM, kDM, 1.0f, 0);

  // 4. band + global-column attention
  const int bandSmem =
      32 * kL * 4 + 64 * kL * 2 + 32 * kL * 2 + 64 * 32 * 2 + kL * 4;
  band_attn_kernel<<<kB * kH * kNC * 4, 64, bandSmem, stream>>>(
      qhb, khb, vhb, pad, gm, gidx, attnf);

  // 5. global-query attention (overwrites gidx rows)
  global_attn_kernel<<<kB * kH, 128, 0, stream>>>(qhb, khb, vhb, pad, gidx,
                                                  attnf);

  // 6. output projection + LN1
  cvt(attnf, attnh, nBSdm);
  gemm_f16_kernel<<<gP, 128, 0, stream>>>(attnh, woh, bo, of, nullptr,
                                          Mrows, kDM, kDM, 1.0f, 0);
  add_ln_kernel<<<Mrows, 256, 0, stream>>>(x, of, g1, be1, h1f, h1h);

  // 7. FFN
  const dim3 gF1(kDFF / 64, Mrows / 64);
  gemm_f16_kernel<<<gF1, 128, 0, stream>>>(h1h, w1h, b1, nullptr, ff1h,
                                           Mrows, kDFF, kDM, 1.0f, 1);
  gemm_f16_kernel<<<gP, 128, 0, stream>>>(ff1h, w2h, b2, ff2f, nullptr,
                                          Mrows, kDM, kDFF, 1.0f, 0);

  // 8. final residual + LN -> d_out
  add_ln_kernel<<<Mrows, 256, 0, stream>>>(h1f, ff2f, g2, be2, out, nullptr);
}
